// PixelContrastLoss_19250043420777
// MI455X (gfx1250) — matrix operand order repaired and checked by reading, not compiled, not run on hardware
//
#include <hip/hip_runtime.h>
#include <hip/hip_bf16.h>

// ---------------------------------------------------------------------------
// PixelContrastLoss for MI455X (gfx1250, wave32, WMMA f16, f32 accumulate).
//
// Shift-invariant reformulation (dot <= 1 for unit vectors):
//   l_ij    = (dot_ij - 1)/T          (computed in log2 units: l2 = fma(dot,C1,-C1))
//   total_i = sum_j exp(l_ij)                        (pass 1: big WMMA GEMM)
//   posE_i  = sum_{j in class(y_i)} exp(l_ij)        (pass 2a)
//   posL_i  = sum_{j in class(y_i)} l_ij             (pass 2a, log2 units)
//   neg_i   = total_i - posE_i                       (finalize 1)
//   Slog_i  = sum_{pos} log(exp(l_ij)+neg_i)         (pass 2b, log2 units)
//   loss    = mean_i [ -(posL_i - Slog_i)*ln2 / (5000 + 1e-4) ]
//
// Pass 1 waves keep 64 rows of A resident in 256 VGPRs (f16) -> B bytes/MAC
// drops 4x vs a 16-row tile; Q/X are pre-converted to f16 in workspace so the
// inner loop is pure global_load_b128 + v_wmma (no v_cvt) when ws permits.
// ---------------------------------------------------------------------------

typedef __attribute__((ext_vector_type(16))) _Float16 v16h;
typedef __attribute__((ext_vector_type(8)))  _Float16 h8;
typedef __attribute__((ext_vector_type(4)))  _Float16 h4;
typedef __attribute__((ext_vector_type(8)))  float    v8f;
typedef __attribute__((ext_vector_type(4)))  float    f32x4;

#define N_ROWS   1020      // 10 views * 102 anchors
#define N_COLS   95000     // 19 classes * 5000
#define CACHE    5000
#define NCLASS   19
#define KDIM     256
// C1 = (1/T) * log2(e);  final scale by ln2.
#define C1_LOG2  20.609929792747908f
#define LN2F     0.6931471805599453f

// ---------------- fragment loaders --------------------------------------
__device__ inline v16h pack4(f32x4 a, f32x4 b, f32x4 c, f32x4 d) {
  v16h h;
#pragma unroll
  for (int i = 0; i < 4; ++i) {
    h[i] = (_Float16)a[i]; h[4+i] = (_Float16)b[i];
    h[8+i] = (_Float16)c[i]; h[12+i] = (_Float16)d[i];
  }
  return h;
}
// fp32 source
__device__ inline v16h fragB_f32(const float* p) {          // 16 contiguous
  return pack4(*(const f32x4*)p, *(const f32x4*)(p+4),
               *(const f32x4*)(p+8), *(const f32x4*)(p+12));
}
__device__ inline v16h fragA_f32(const float* p) {          // 16 contiguous
  return pack4(*(const f32x4*)p, *(const f32x4*)(p+4),
               *(const f32x4*)(p+8), *(const f32x4*)(p+12));
}
// f16 source
__device__ inline v16h fragB_f16(const _Float16* p) {       // 16 contiguous halves
  h8 a = *(const h8*)p, b = *(const h8*)(p + 8);
  v16h h;
#pragma unroll
  for (int i = 0; i < 8; ++i) { h[i] = a[i]; h[8+i] = b[i]; }
  return h;
}
__device__ inline v16h fragA_f16(const _Float16* p) {       // 16 contiguous halves
  h8 a = *(const h8*)p, b = *(const h8*)(p + 8);
  v16h h;
#pragma unroll
  for (int i = 0; i < 8; ++i) { h[i] = a[i]; h[8+i] = b[i]; }
  return h;
}

// anchor_feature row i = X_anchor[i % 102][i / 102][:]
__device__ inline const float* anchor_row(const float* X, int i) {
  return X + (size_t)((i % 102) * 10 + (i / 102)) * KDIM;
}

// ---------------- f32 -> f16 pre-conversion ------------------------------
__global__ __launch_bounds__(256)
void cvt_q(const float* __restrict__ Q, _Float16* __restrict__ Qh, int n4) {
  int i = blockIdx.x * 256 + threadIdx.x;
  if (i < n4) {
    f32x4 v = ((const f32x4*)Q)[i];
    h4 h;
#pragma unroll
    for (int j = 0; j < 4; ++j) h[j] = (_Float16)v[j];
    ((h4*)Qh)[i] = h;
  }
}
__global__ __launch_bounds__(256)
void cvt_x(const float* __restrict__ X, _Float16* __restrict__ Xh) {
  int t = blockIdx.x * 256 + threadIdx.x;       // N_ROWS*64 threads, 4 elems each
  if (t < N_ROWS * 64) {
    int row = t >> 6, k4 = (t & 63) * 4;
    f32x4 v = *(const f32x4*)(anchor_row(X, row) + k4);
    h4 h;
#pragma unroll
    for (int j = 0; j < 4; ++j) h[j] = (_Float16)v[j];
    *(h4*)(Xh + (size_t)row * KDIM + k4) = h;
  }
}

// ---------------------------------------------------------------------------
// Pass 1: total_i over all 95000 columns. One wave = 64 rows (4 row tiles).
// ---------------------------------------------------------------------------
template <bool F16>
__global__ __launch_bounds__(32)
void pass1_total_k(const float* __restrict__ X, const float* __restrict__ Q,
                   const _Float16* __restrict__ Xh, const _Float16* __restrict__ Qh,
                   float* __restrict__ total) {
  const int lane = threadIdx.x;
  const int g  = lane >> 4;
  const int ln = lane & 15;
  const int r0 = blockIdx.x * 64;

  v16h af[4][8];
#pragma unroll
  for (int rt = 0; rt < 4; ++rt) {
    int row = r0 + rt * 16 + ln; if (row > N_ROWS - 1) row = N_ROWS - 1;
    if (F16) {
      const _Float16* p = Xh + (size_t)row * KDIM + 16 * g;
#pragma unroll
      for (int it = 0; it < 8; ++it) af[rt][it] = fragA_f16(p + it * 32);
    } else {
      const float* p = anchor_row(X, row) + 16 * g;
#pragma unroll
      for (int it = 0; it < 8; ++it) af[rt][it] = fragA_f32(p + it * 32);
    }
  }

  float acc[4][8];
#pragma unroll
  for (int rt = 0; rt < 4; ++rt)
#pragma unroll
    for (int v = 0; v < 8; ++v) acc[rt][v] = 0.f;

  const int NT = (N_COLS + 15) / 16;   // 5938
  for (int ct = blockIdx.y; ct < NT; ct += gridDim.y) {
    int j = ct * 16 + ln;
    const bool cv = (j < N_COLS);
    if (j > N_COLS - 1) j = N_COLS - 1;

    v16h bf[8];
    if (F16) {
      const _Float16* p = Qh + (size_t)j * KDIM + 16 * g;
#pragma unroll
      for (int it = 0; it < 8; ++it) bf[it] = fragB_f16(p + it * 32);
    } else {
      const float* p = Q + (size_t)j * KDIM + 16 * g;
#pragma unroll
      for (int it = 0; it < 8; ++it) bf[it] = fragB_f32(p + it * 32);
    }

    v8f cf[4] = {};
#pragma unroll
    for (int it = 0; it < 8; ++it)
#pragma unroll
      for (int rt = 0; rt < 4; ++rt)   // 4 independent accumulation chains
        cf[rt] = __builtin_amdgcn_wmma_f32_16x16x32_f16(
                    false, af[rt][it], false, bf[it], (short)0, cf[rt], false, false);

#pragma unroll
    for (int rt = 0; rt < 4; ++rt)
#pragma unroll
      for (int v = 0; v < 8; ++v) {
        float e = exp2f(fmaf(cf[rt][v], C1_LOG2, -C1_LOG2));
        if (cv) acc[rt][v] += e;
      }
  }

#pragma unroll
  for (int rt = 0; rt < 4; ++rt)
#pragma unroll
    for (int v = 0; v < 8; ++v) {
#pragma unroll
      for (int m = 1; m < 16; m <<= 1) acc[rt][v] += __shfl_xor(acc[rt][v], m, 32);
    }
  if (ln == 0) {
#pragma unroll
    for (int rt = 0; rt < 4; ++rt)
#pragma unroll
      for (int v = 0; v < 8; ++v) {
        int row = r0 + rt * 16 + 8 * g + v;
        if (row < N_ROWS) atomicAdd(&total[row], acc[rt][v]);
      }
  }
}

// ---------------------------------------------------------------------------
// Pass 2: positive class blocks only (rows grouped by class via perm[]).
//   mode 0: posE += exp(l), posL += l2        mode 1: slog += log2(exp(l)+neg)
// ---------------------------------------------------------------------------
template <bool F16>
__global__ __launch_bounds__(32)
void pass2_pos_k(const float* __restrict__ X, const float* __restrict__ Q,
                 const _Float16* __restrict__ Xh, const _Float16* __restrict__ Qh,
                 const int* __restrict__ perm, const int* __restrict__ cstart,
                 const float* __restrict__ negsum,
                 float* __restrict__ posE, float* __restrict__ posL,
                 float* __restrict__ slog, int mode) {
  const int lane = threadIdx.x;
  const int g  = lane >> 4;
  const int ln = lane & 15;
  const int c  = blockIdx.x;
  const int t  = blockIdx.y;
  const int base = cstart[c];
  const int cnt  = cstart[c + 1] - base;
  if (16 * t >= cnt) return;            // uniform per block

  int rseg = 16 * t + ln; if (rseg > cnt - 1) rseg = cnt - 1;
  const int orig = perm[base + rseg];
  v16h af[8];
  if (F16) {
    const _Float16* p = Xh + (size_t)orig * KDIM + 16 * g;
#pragma unroll
    for (int it = 0; it < 8; ++it) af[it] = fragA_f16(p + it * 32);
  } else {
    const float* p = anchor_row(X, orig) + 16 * g;
#pragma unroll
    for (int it = 0; it < 8; ++it) af[it] = fragA_f32(p + it * 32);
  }

  int rowsOrig[8]; float ns[8];
#pragma unroll
  for (int v = 0; v < 8; ++v) {
    int rl = 16 * t + 8 * g + v;
    int rc = rl > cnt - 1 ? cnt - 1 : rl;
    rowsOrig[v] = perm[base + rc];
    ns[v] = (mode == 1) ? negsum[rowsOrig[v]] : 0.f;
  }

  float aE[8], aL[8], aG[8];
#pragma unroll
  for (int v = 0; v < 8; ++v) { aE[v] = 0.f; aL[v] = 0.f; aG[v] = 0.f; }

  const int NT = (CACHE + 15) / 16;     // 313
  for (int ct = blockIdx.z; ct < NT; ct += gridDim.z) {
    int jc = ct * 16 + ln;
    const bool cv = (jc < CACHE);
    if (jc > CACHE - 1) jc = CACHE - 1;
    const size_t j = (size_t)(c * CACHE + jc);

    v16h bf[8];
    if (F16) {
      const _Float16* p = Qh + j * KDIM + 16 * g;
#pragma unroll
      for (int it = 0; it < 8; ++it) bf[it] = fragB_f16(p + it * 32);
    } else {
      const float* p = Q + j * KDIM + 16 * g;
#pragma unroll
      for (int it = 0; it < 8; ++it) bf[it] = fragB_f32(p + it * 32);
    }

    v8f cf = {};
#pragma unroll
    for (int it = 0; it < 8; ++it)
      cf = __builtin_amdgcn_wmma_f32_16x16x32_f16(
              false, af[it], false, bf[it], (short)0, cf, false, false);

#pragma unroll
    for (int v = 0; v < 8; ++v) {
      float l2 = fmaf(cf[v], C1_LOG2, -C1_LOG2);
      float e  = exp2f(l2);
      if (cv) {
        if (mode == 0) { aE[v] += e; aL[v] += l2; }
        else           { aG[v] += log2f(e + ns[v]); }
      }
    }
  }

#pragma unroll
  for (int v = 0; v < 8; ++v) {
#pragma unroll
    for (int m = 1; m < 16; m <<= 1) {
      aE[v] += __shfl_xor(aE[v], m, 32);
      aL[v] += __shfl_xor(aL[v], m, 32);
      aG[v] += __shfl_xor(aG[v], m, 32);
    }
  }
  if (ln == 0) {
#pragma unroll
    for (int v = 0; v < 8; ++v) {
      int rl = 16 * t + 8 * g + v;
      if (rl < cnt) {
        int o = rowsOrig[v];
        if (mode == 0) { atomicAdd(&posE[o], aE[v]); atomicAdd(&posL[o], aL[v]); }
        else           { atomicAdd(&slog[o], aG[v]); }
      }
    }
  }
}

// ---------------------------------------------------------------------------
// Small helper kernels
// ---------------------------------------------------------------------------
__global__ void zero_ws(float* p, int n) {
  int i = blockIdx.x * blockDim.x + threadIdx.x;
  if (i < n) p[i] = 0.f;
}

__global__ __launch_bounds__(1024)
void build_perm(const int* __restrict__ y, int* __restrict__ perm,
                int* __restrict__ cstart) {
  __shared__ int cnt[NCLASS];
  __shared__ int off[NCLASS + 1];
  __shared__ int cur[NCLASS];
  const int tid = threadIdx.x;
  if (tid < NCLASS) cnt[tid] = 0;
  __syncthreads();
  int yi = 0;
  if (tid < N_ROWS) { yi = y[tid % 102]; atomicAdd(&cnt[yi], 1); }
  __syncthreads();
  if (tid == 0) {
    off[0] = 0;
    for (int c = 0; c < NCLASS; ++c) { off[c + 1] = off[c] + cnt[c]; cur[c] = off[c]; }
  }
  __syncthreads();
  if (tid < N_ROWS) { int s = atomicAdd(&cur[yi], 1); perm[s] = tid; }
  if (tid < NCLASS + 1) cstart[tid] = off[tid];
}

__global__ void finalize1(const float* __restrict__ total,
                          const float* __restrict__ posE,
                          float* __restrict__ negsum, int n) {
  int i = blockIdx.x * blockDim.x + threadIdx.x;
  if (i < n) negsum[i] = total[i] - posE[i];
}

__global__ __launch_bounds__(1024)
void finalize2(const float* __restrict__ posL, const float* __restrict__ slog,
               float* __restrict__ out, int n) {
  __shared__ float red[1024];
  const int tid = threadIdx.x;
  const float scale = LN2F / (5000.0f + 1.0e-4f);
  float v = 0.f;
  for (int i = tid; i < n; i += 1024) v += -(posL[i] - slog[i]) * scale;
  red[tid] = v; __syncthreads();
  for (int s = 512; s > 0; s >>= 1) {
    if (tid < s) red[tid] += red[tid + s];
    __syncthreads();
  }
  if (tid == 0) out[0] = red[0] / (float)n;
}

// ---------------------------------------------------------------------------
extern "C" void kernel_launch(void* const* d_in, const int* in_sizes, int n_in,
                              void* d_out, int out_size, void* d_ws, size_t ws_size,
                              hipStream_t stream) {
  const float* X = (const float*)d_in[0];   // [102,10,256] fp32
  const int*   y = (const int*)d_in[1];     // [102] int32
  const float* Q = (const float*)d_in[2];   // [19,5000,256] fp32

  float* ws     = (float*)d_ws;
  float* total  = ws;                 // [1024]
  float* posE   = ws + 1024;          // [1024]
  float* posL   = ws + 2048;          // [1024]
  float* slog   = ws + 3072;          // [1024]
  float* negsum = ws + 4096;          // [1024]
  int*   perm   = (int*)(ws + 5120);  // [1024]
  int*   cstart = perm + 1024;        // [20]

  const size_t qhBytes = (size_t)N_COLS * KDIM * 2;          // 48.64 MB
  const size_t xhBytes = (size_t)N_ROWS * KDIM * 2;
  const size_t f16Need = 32768 + qhBytes + xhBytes + 256;
  const bool useF16 = (ws_size >= f16Need);
  _Float16* Qh = (_Float16*)((char*)d_ws + 32768);
  _Float16* Xh = (_Float16*)((char*)d_ws + 32768 + qhBytes);

  zero_ws<<<(4096 + 255) / 256, 256, 0, stream>>>(ws, 4096);
  build_perm<<<1, 1024, 0, stream>>>(y, perm, cstart);

  if (useF16) {
    const int n4 = N_COLS * KDIM / 4;
    cvt_q<<<(n4 + 255) / 256, 256, 0, stream>>>(Q, Qh, n4);
    cvt_x<<<(N_ROWS * 64 + 255) / 256, 256, 0, stream>>>(X, Xh);

    pass1_total_k<true><<<dim3(16, 512), 32, 0, stream>>>(X, Q, Xh, Qh, total);
    pass2_pos_k<true><<<dim3(NCLASS, 64, 16), 32, 0, stream>>>(
        X, Q, Xh, Qh, perm, cstart, negsum, posE, posL, slog, 0);
    finalize1<<<(N_ROWS + 255) / 256, 256, 0, stream>>>(total, posE, negsum, N_ROWS);
    pass2_pos_k<true><<<dim3(NCLASS, 64, 16), 32, 0, stream>>>(
        X, Q, Xh, Qh, perm, cstart, negsum, posE, posL, slog, 1);
  } else {
    pass1_total_k<false><<<dim3(16, 512), 32, 0, stream>>>(X, Q, nullptr, nullptr, total);
    pass2_pos_k<false><<<dim3(NCLASS, 64, 16), 32, 0, stream>>>(
        X, Q, nullptr, nullptr, perm, cstart, negsum, posE, posL, slog, 0);
    finalize1<<<(N_ROWS + 255) / 256, 256, 0, stream>>>(total, posE, negsum, N_ROWS);
    pass2_pos_k<false><<<dim3(NCLASS, 64, 16), 32, 0, stream>>>(
        X, Q, nullptr, nullptr, perm, cstart, negsum, posE, posL, slog, 1);
  }

  finalize2<<<1, 1024, 0, stream>>>(posL, slog, (float*)d_out, N_ROWS);
}
